// BidirectionalAttention_3865470566811
// MI455X (gfx1250) — compile-verified, matrix operand order checked by reading, not run on hardware
//
#include <hip/hip_runtime.h>
#include <hip/hip_bf16.h>
#include <math.h>

typedef __attribute__((ext_vector_type(16))) _Float16 v16h;
typedef __attribute__((ext_vector_type(8)))  _Float16 v8h;
typedef __attribute__((ext_vector_type(8)))  float    v8f;
typedef __attribute__((ext_vector_type(4)))  float    v4f;

#define Bn   8
#define S    2048
#define KD   256
#define AD   128
#define VD   256

// ---------------------------------------------------------------------------
// WMMA helper: D = A(16x32 f16) * B(32x16 f16) + C(16x16 f32)
// ---------------------------------------------------------------------------
__device__ __forceinline__ v8f wmma_f16(v16h a, v16h b, v8f c) {
  return __builtin_amdgcn_wmma_f32_16x16x32_f16(
      /*neg_a=*/false, a, /*neg_b=*/false, b,
      /*c_mod=*/(short)0, c, /*reuse_a=*/false, /*reuse_b=*/false);
}

// A-fragment (16x32, f16) loaded from a row-major f16 matrix.
// ISA 16-bit A layout: lane<16 -> row=m0+lane, elems 0..7 = K kb..kb+7,
// elems 8..15 = K kb+16..kb+23 with kb=0; lane>=16 -> same row, kb=8.
__device__ __forceinline__ v16h afrag_f16(const _Float16* __restrict__ A,
                                          int ld, int m0, int k0) {
  const int lane = threadIdx.x & 31;
  const int row  = m0 + (lane & 15);
  const int kb   = k0 + ((lane >> 4) << 3);
  const _Float16* p = A + (size_t)row * ld + kb;
  v8h lo = *(const v8h*)p;
  v8h hi = *(const v8h*)(p + 16);
  return __builtin_shufflevector(lo, hi, 0,1,2,3,4,5,6,7,8,9,10,11,12,13,14,15);
}

// Same fragment, but source matrix is f32 (convert on the fly).
__device__ __forceinline__ v16h afrag_f32(const float* __restrict__ A,
                                          int ld, int m0, int k0) {
  const int lane = threadIdx.x & 31;
  const int row  = m0 + (lane & 15);
  const int kb   = k0 + ((lane >> 4) << 3);
  const float* p = A + (size_t)row * ld + kb;
  const v4f* q = (const v4f*)p;
  v4f a0 = q[0], a1 = q[1];      // K kb .. kb+7
  v4f a2 = q[4], a3 = q[5];      // K kb+16 .. kb+23
  v16h r;
#pragma unroll
  for (int e = 0; e < 4; ++e) {
    r[e]      = (_Float16)a0[e];
    r[4 + e]  = (_Float16)a1[e];
    r[8 + e]  = (_Float16)a2[e];
    r[12 + e] = (_Float16)a3[e];
  }
  return r;
}

// K-offset inside a 32-wide K-tile for fragment element e of a given lane.
__device__ __forceinline__ int frag_koff(int lane, int e) {
  return ((lane >> 4) << 3) + (e & 7) + ((e >> 3) << 4);
}

// ---------------------------------------------------------------------------
// Kernel 1: pre-swizzle V (f32 [B,S,VD]) into B-fragment layout, f16.
// Layout: [b][ktile(64)][ntile(16)][lane(32)][elem(16)] halves.
// ---------------------------------------------------------------------------
__global__ __launch_bounds__(256) void vfrag_prep(const float* __restrict__ V,
                                                  _Float16* __restrict__ F) {
  const size_t f = (size_t)blockIdx.x * 256 + threadIdx.x;   // lane-frag id
  const int lane  = (int)(f & 31);
  const int ntile = (int)((f >> 5) & 15);
  const int ktile = (int)((f >> 9) & 63);
  const int b     = (int)(f >> 15);
  const float* Vb = V + (size_t)b * S * VD;
  const int n  = ntile * 16 + (lane & 15);
  const int k0 = ktile * 32;
  v16h r;
#pragma unroll
  for (int e = 0; e < 16; ++e)
    r[e] = (_Float16)Vb[(size_t)(k0 + frag_koff(lane, e)) * VD + n];
  *(v16h*)(F + (f << 4)) = r;
}

// B-fragment fetch from pre-swizzled buffer: one contiguous 32B load per lane.
__device__ __forceinline__ v16h bfrag_pre(const _Float16* __restrict__ F,
                                          int ktile, int ntile) {
  const int lane = threadIdx.x & 31;
  return *(const v16h*)(F + (((size_t)(ktile * 16 + ntile) * 32 + lane) << 4));
}

// ---------------------------------------------------------------------------
// Kernel 2: projection  P = Kmat @ W + bias  (f32 in, f16 out [rows, AD]).
// Block = 256 thr (8 waves) -> 128x128 tile (full AD). Wk staged in LDS
// in fragment layout (8 ktiles * 8 ntiles * 32 lanes * 16 halves = 64KB).
// ---------------------------------------------------------------------------
__global__ __launch_bounds__(256) void proj_kernel(const float* __restrict__ Kmat,
                                                   const float* __restrict__ W,
                                                   const float* __restrict__ bias,
                                                   _Float16* __restrict__ P) {
  __shared__ __align__(32) _Float16 wfrag[8 * 8 * 32 * 16];
  const int t = threadIdx.x;
  // Stage Wk fragments: 2048 lane-frags, 8 per thread.
#pragma unroll
  for (int s = 0; s < 8; ++s) {
    const int f    = t * 8 + s;
    const int lane = f & 31;
    const int nt   = (f >> 5) & 7;
    const int kt   = f >> 8;
    const int n    = nt * 16 + (lane & 15);
    const int k0   = kt * 32;
    _Float16* dst  = &wfrag[f * 16];
#pragma unroll
    for (int e = 0; e < 16; ++e)
      dst[e] = (_Float16)W[(size_t)(k0 + frag_koff(lane, e)) * AD + n];
  }
  __syncthreads();

  const int wave = t >> 5, lane = t & 31;
  const int m0 = blockIdx.x * 128 + wave * 16;
  v8f acc[8] = {};
#pragma unroll
  for (int kt = 0; kt < 8; ++kt) {
    v16h a = afrag_f32(Kmat, KD, m0, kt * 32);
#pragma unroll
    for (int nt = 0; nt < 8; ++nt) {
      v16h b = *(const v16h*)&wfrag[((kt * 8 + nt) * 32 + lane) * 16];
      acc[nt] = wmma_f16(a, b, acc[nt]);
    }
  }
  // Epilogue: + bias, convert to f16, store.
#pragma unroll
  for (int nt = 0; nt < 8; ++nt) {
    const int col = nt * 16 + (lane & 15);
    const float bv = bias[col];
#pragma unroll
    for (int r = 0; r < 8; ++r) {
      const int row = m0 + r + ((lane >> 4) << 3);
      P[(size_t)row * AD + col] = (_Float16)(acc[nt][r] + bv);
    }
  }
}

// ---------------------------------------------------------------------------
// Kernel 3: score[b,i,j] = sum_a p1[b,i,a]*p2[b,j,a], masked (XOR) -> -inf.
// Both operands row-major [S, AD] f16: B-fragment load == A-fragment load
// of the p2 row block. Block 256 thr -> 128x128 tile, K=128 in 4 steps.
// ---------------------------------------------------------------------------
__global__ __launch_bounds__(256) void score_kernel(const _Float16* __restrict__ P1,
                                                    const _Float16* __restrict__ P2,
                                                    float* __restrict__ Sc,
                                                    const long long* __restrict__ len1,
                                                    const long long* __restrict__ len2) {
  const int b = blockIdx.z;
  const int i0 = blockIdx.x * 128;
  const int j0 = blockIdx.y * 128;
  const int t = threadIdx.x, wave = t >> 5, lane = t & 31;
  const _Float16* p1 = P1 + (size_t)b * S * AD;
  const _Float16* p2 = P2 + (size_t)b * S * AD;
  const int m0 = i0 + wave * 16;
  v8f acc[8] = {};
#pragma unroll
  for (int kt = 0; kt < 4; ++kt) {
    v16h a = afrag_f16(p1, AD, m0, kt * 32);
#pragma unroll
    for (int nt = 0; nt < 8; ++nt) {
      v16h bf = afrag_f16(p2, AD, j0 + nt * 16, kt * 32);
      acc[nt] = wmma_f16(a, bf, acc[nt]);
    }
  }
  const long long L1 = len1[b], L2 = len2[b];
  float* sb = Sc + (size_t)b * S * S;
#pragma unroll
  for (int nt = 0; nt < 8; ++nt) {
    const int col = j0 + nt * 16 + (lane & 15);
    const bool m2 = col < L2;
#pragma unroll
    for (int r = 0; r < 8; ++r) {
      const int row = m0 + r + ((lane >> 4) << 3);
      const bool m1 = row < L1;
      float v = acc[nt][r];
      if (m1 != m2) v = -__builtin_inff();
      sb[(size_t)row * S + col] = v;
    }
  }
}

// ---------------------------------------------------------------------------
// Kernel 4: row softmax -> w2. One 256-thread block per (b,i) row of 2048.
// ---------------------------------------------------------------------------
__global__ __launch_bounds__(256) void row_softmax(const float* __restrict__ Sc,
                                                   float* __restrict__ W2) {
  const size_t row = blockIdx.x;            // b*S + i
  const float* src = Sc + row * S;
  float* dst = W2 + row * S;
  const int t = threadIdx.x;
  __shared__ float red[256];
  float vals[8];
  float mx = -__builtin_inff();
#pragma unroll
  for (int r = 0; r < 8; ++r) { vals[r] = src[t + 256 * r]; mx = fmaxf(mx, vals[r]); }
  red[t] = mx; __syncthreads();
  for (int s = 128; s > 0; s >>= 1) { if (t < s) red[t] = fmaxf(red[t], red[t + s]); __syncthreads(); }
  mx = red[0]; __syncthreads();
  float sum = 0.f;
#pragma unroll
  for (int r = 0; r < 8; ++r) { vals[r] = __expf(vals[r] - mx); sum += vals[r]; }
  red[t] = sum; __syncthreads();
  for (int s = 128; s > 0; s >>= 1) { if (t < s) red[t] += red[t + s]; __syncthreads(); }
  const float inv = 1.0f / red[0];
#pragma unroll
  for (int r = 0; r < 8; ++r) dst[t + 256 * r] = vals[r] * inv;
}

// ---------------------------------------------------------------------------
// Kernel 5: column softmax stats (max, sum of exp) per (b, j).
// Block = 1024 thr: 128 columns x 8 i-strips, fully coalesced reads.
// ---------------------------------------------------------------------------
__global__ __launch_bounds__(1024) void col_stats(const float* __restrict__ Sc,
                                                  float* __restrict__ cmax,
                                                  float* __restrict__ csum) {
  const int blk = blockIdx.x;
  const int b = blk >> 4;                  // 16 j-blocks of 128 per batch
  const int j0 = (blk & 15) * 128;
  const int t = threadIdx.x;
  const int jl = t & 127, strip = t >> 7;  // strip in 0..7
  const float* base = Sc + (size_t)b * S * S + j0;
  __shared__ float smax[8][128];
  __shared__ float ssum[8][128];
  float mx = -__builtin_inff();
  for (int i = strip; i < S; i += 8) mx = fmaxf(mx, base[(size_t)i * S + jl]);
  smax[strip][jl] = mx; __syncthreads();
  if (strip == 0) {
#pragma unroll
    for (int s = 1; s < 8; ++s) mx = fmaxf(mx, smax[s][jl]);
    smax[0][jl] = mx;
  }
  __syncthreads();
  mx = smax[0][jl];                        // finite by construction of mask
  float sm = 0.f;
  for (int i = strip; i < S; i += 8) sm += __expf(base[(size_t)i * S + jl] - mx);
  ssum[strip][jl] = sm; __syncthreads();
  if (strip == 0) {
#pragma unroll
    for (int s = 1; s < 8; ++s) sm += ssum[s][jl];
    cmax[(size_t)b * S + j0 + jl] = mx;
    csum[(size_t)b * S + j0 + jl] = sm;
  }
}

// ---------------------------------------------------------------------------
// Kernel 6: w1[b,j,i] = exp(score[b,i,j]-cmax[b,j])/csum[b,j], via 64x64
// LDS transpose tile so both read and write are coalesced.
// ---------------------------------------------------------------------------
__global__ __launch_bounds__(256) void w1_transpose(const float* __restrict__ Sc,
                                                    const float* __restrict__ cmax,
                                                    const float* __restrict__ csum,
                                                    float* __restrict__ W1) {
  const int b = blockIdx.z;
  const int i0 = blockIdx.x * 64;
  const int j0 = blockIdx.y * 64;
  __shared__ float tile[64][65];
  const int t = threadIdx.x;
  const float* sb = Sc + (size_t)b * S * S;
  float* wb = W1 + (size_t)b * S * S;
  {
    const int jl = t & 63, ib = t >> 6;    // 4 i-rows in flight
    const float cm = cmax[(size_t)b * S + j0 + jl];
    const float inv = 1.0f / csum[(size_t)b * S + j0 + jl];
#pragma unroll
    for (int r = 0; r < 16; ++r) {
      const int il = ib + r * 4;
      const float v = sb[(size_t)(i0 + il) * S + (j0 + jl)];
      tile[jl][il] = __expf(v - cm) * inv;
    }
  }
  __syncthreads();
  {
    const int il = t & 63, jb = t >> 6;
#pragma unroll
    for (int r = 0; r < 16; ++r) {
      const int jl = jb + r * 4;
      wb[(size_t)(j0 + jl) * S + (i0 + il)] = tile[jl][il];
    }
  }
}

// ---------------------------------------------------------------------------
// Kernel 7: O = Wmat(f32 [B,S,S]) @ Vfrag(f16), rows masked by len.
// Grid (S/128, VD/128, B); block 256 = 8 waves; 64 K-steps of 32.
// ---------------------------------------------------------------------------
__global__ __launch_bounds__(256) void ogemm_kernel(const float* __restrict__ Wmat,
                                                    const _Float16* __restrict__ Vfrag,
                                                    float* __restrict__ O,
                                                    const long long* __restrict__ lens) {
  const int b = blockIdx.z;
  const int t = threadIdx.x, wave = t >> 5, lane = t & 31;
  const int m0 = blockIdx.x * 128 + wave * 16;
  const int nblk = blockIdx.y;             // 0..1 (128 cols each)
  const float* Wb = Wmat + (size_t)b * S * S;
  const _Float16* Vf = Vfrag + (size_t)b * (S / 32) * 16 * 32 * 16;
  v8f acc[8] = {};
  for (int kt = 0; kt < S / 32; ++kt) {
    v16h a = afrag_f32(Wb, S, m0, kt * 32);
#pragma unroll
    for (int nt = 0; nt < 8; ++nt)
      acc[nt] = wmma_f16(a, bfrag_pre(Vf, kt, nblk * 8 + nt), acc[nt]);
  }
  const long long L = lens[b];
  float* Ob = O + (size_t)b * S * VD;
#pragma unroll
  for (int nt = 0; nt < 8; ++nt) {
    const int col = nblk * 128 + nt * 16 + (lane & 15);
#pragma unroll
    for (int r = 0; r < 8; ++r) {
      const int row = m0 + r + ((lane >> 4) << 3);
      Ob[(size_t)row * VD + col] = (row < L) ? acc[nt][r] : 0.0f;
    }
  }
}

// ---------------------------------------------------------------------------
// Launch
// ---------------------------------------------------------------------------
extern "C" void kernel_launch(void* const* d_in, const int* in_sizes, int n_in,
                              void* d_out, int out_size, void* d_ws, size_t ws_size,
                              hipStream_t stream) {
  (void)in_sizes; (void)n_in; (void)out_size; (void)ws_size;
  const float* k1  = (const float*)d_in[0];
  const float* k2  = (const float*)d_in[1];
  const float* v1  = (const float*)d_in[2];
  const float* v2  = (const float*)d_in[3];
  const float* Wk1 = (const float*)d_in[4];
  const float* bk1 = (const float*)d_in[5];
  const float* Wk2 = (const float*)d_in[6];
  const float* bk2 = (const float*)d_in[7];
  const long long* len1 = (const long long*)d_in[8];
  const long long* len2 = (const long long*)d_in[9];

  float* out = (float*)d_out;
  float* o1    = out;
  float* o2    = o1 + (size_t)Bn * S * VD;
  float* w1    = o2 + (size_t)Bn * S * VD;
  float* w2    = w1 + (size_t)Bn * S * S;
  float* score = w2 + (size_t)Bn * S * S;

  char* ws = (char*)d_ws;
  _Float16* p1  = (_Float16*)ws; ws += (size_t)Bn * S * AD * 2;       // 4 MB
  _Float16* p2  = (_Float16*)ws; ws += (size_t)Bn * S * AD * 2;       // 4 MB
  _Float16* v1f = (_Float16*)ws; ws += (size_t)Bn * S * VD * 2;       // 8 MB
  _Float16* v2f = (_Float16*)ws; ws += (size_t)Bn * S * VD * 2;       // 8 MB
  float* cmax = (float*)ws;      ws += (size_t)Bn * S * 4;
  float* csum = (float*)ws;

  // 1. V fragment pre-swizzle (f32 -> f16 WMMA B-fragments).
  vfrag_prep<<<2048, 256, 0, stream>>>(v1, v1f);
  vfrag_prep<<<2048, 256, 0, stream>>>(v2, v2f);

  // 2. Projections p1/p2 (f16).
  proj_kernel<<<(Bn * S) / 128, 256, 0, stream>>>(k1, Wk1, bk1, p1);
  proj_kernel<<<(Bn * S) / 128, 256, 0, stream>>>(k2, Wk2, bk2, p2);

  // 3. Score + mask.
  score_kernel<<<dim3(S / 128, S / 128, Bn), 256, 0, stream>>>(p1, p2, score, len1, len2);

  // 4. Row softmax -> w2.
  row_softmax<<<Bn * S, 256, 0, stream>>>(score, w2);

  // 5-6. Column softmax -> w1 (stats, then normalize+transpose).
  col_stats<<<Bn * (S / 128), 1024, 0, stream>>>(score, cmax, csum);
  w1_transpose<<<dim3(S / 64, S / 64, Bn), 256, 0, stream>>>(score, cmax, csum, w1);

  // 7. o1 = w1 @ v1 (mask len2 rows), o2 = w2 @ v2 (mask len1 rows).
  ogemm_kernel<<<dim3(S / 128, VD / 128, Bn), 256, 0, stream>>>(w1, v1f, o1, len2);
  ogemm_kernel<<<dim3(S / 128, VD / 128, Bn), 256, 0, stream>>>(w2, v2f, o2, len1);
}
